// MultiStepDecoder_1116691497583
// MI455X (gfx1250) — matrix operand order repaired and checked by reading, not compiled, not run on hardware
//
#include <hip/hip_runtime.h>
#include <hip/hip_bf16.h>

// ---------------- problem constants ----------------
#define BB    32
#define NN    128
#define DD    64
#define HH    128
#define MM    128
#define NHH   256
#define TT    2
#define EE    16256
#define EHALF 8128
#define ET    64      // edges per tile (8128 = 127 * 64, exact)
#define NTILE 127     // tiles per half
#define CHUNKS 16     // chunks per half
#define TPC   8       // tiles per chunk (ceil(127/16))

// byte offset of the TDM staging area inside dynamic LDS (see k_edge layout)
#define SRR_BYTE_OFF 204800
#define SRR_TILE_BYTES (ET * NN * 4)

typedef __bf16 bf16;
typedef __attribute__((ext_vector_type(16))) __bf16 v16bf;
typedef __attribute__((ext_vector_type(8)))  __bf16 v8bf;
typedef __attribute__((ext_vector_type(8)))  float  v8f;
typedef __attribute__((ext_vector_type(4)))  float  v4f;
typedef __attribute__((ext_vector_type(4)))  unsigned int v4u;
typedef __attribute__((ext_vector_type(8)))  int v8i;
typedef __attribute__((ext_vector_type(4)))  int v4i;

static __device__ __forceinline__ v8f zero8() {
  v8f z;
#pragma unroll
  for (int i = 0; i < 8; ++i) z[i] = 0.0f;
  return z;
}

static __device__ __forceinline__ v8f wmma_bf16(v16bf a, v16bf b, v8f c) {
  // D = A*B + C, 16x16x32, fp32 accumulate
  return __builtin_amdgcn_wmma_f32_16x16x32_bf16(false, a, false, b, (short)0, c,
                                                 false, false);
}

// ---- Tensor Data Mover: async DMA of a 64x128 fp32 tile, global -> LDS ----
// D# per CDNA5 ISA:
//  group0: count=1 | lds_addr | global_addr[56:0] | type=2 ("image")
//  group1: data_size=4B; tensor_dim0=128, tensor_dim1=64, tile_dim0=128,
//          tile_dim1=64, tensor_dim0_stride=128  (2-D tile, no padding)
//  groups 2/3: unused (2-D tensor)
static __device__ __forceinline__ void tdm_load_tile_f32(const float* gsrc,
                                                         unsigned ldsByteOff) {
  const unsigned long long ga = (unsigned long long)(size_t)gsrc;
  v4u g0;
  g0[0] = 1u;                                                // count=1
  g0[1] = ldsByteOff;                                        // lds_addr
  g0[2] = (unsigned)ga;                                      // gaddr[31:0]
  g0[3] = (unsigned)((ga >> 32) & 0x01FFFFFFu) | (2u << 30); // gaddr[56:32]|type
  v8i g1;
  g1[0] = 0x20000;               // data_size=4B (bits 17:16=2), no multicast
  g1[1] = (int)(128u << 16);     // tensor_dim0 = 128   ([31:16] = lo16)
  g1[2] = (int)(64u << 16);      // dim0 hi=0 | tensor_dim1 lo16 = 64
  g1[3] = (int)(128u << 16);     // dim1 hi=0 | tile_dim0 = 128
  g1[4] = 64;                    // tile_dim1 = 64, tile_dim2 = 0
  g1[5] = 128;                   // tensor_dim0_stride = 128 (lo32)
  g1[6] = 0;                     // stride0 hi | stride1 lo
  g1[7] = 0;                     // stride1 hi
  v4i z4 = {0, 0, 0, 0};
  v8i z8 = {0, 0, 0, 0, 0, 0, 0, 0};
  __builtin_amdgcn_tensor_load_to_lds(g0, g1, z4, z4, z8, 0);
}

// A fragment (16x32 bf16) from row-major bf16 LDS tile.
// Contiguous-k order used here (matches B): lane L (row = L&15, half = L>>4); elem e -> K = 16*half + e.
// Two 16-byte DS loads per lane.
static __device__ __forceinline__ v16bf frag_a_lds(const bf16* base, int ld,
                                                   int row0, int k0, int lane) {
  const int h = (lane >> 4) & 1, r = lane & 15;
  const bf16* p = base + (row0 + r) * ld + k0 + 16 * h;
  v8bf lo = *(const v8bf*)p;
  v8bf hi = *(const v8bf*)(p + 8);
  v16bf o;
#pragma unroll
  for (int i = 0; i < 8; ++i) { o[i] = lo[i]; o[8 + i] = hi[i]; }
  return o;
}

// B fragment (32x16 bf16) from LDS staged as [col][k] (i.e. transposed).
// Contiguous-k order used here (matches A): lane L (col = L&15, half = L>>4); elem e -> K = 16*half + e.
static __device__ __forceinline__ v16bf frag_b_lds(const bf16* base, int ld,
                                                   int col0, int k0, int lane) {
  const int h = (lane >> 4) & 1, c = lane & 15;
  const bf16* p = base + (col0 + c) * ld + k0 + 16 * h;
  v8bf lo = *(const v8bf*)p;
  v8bf hi = *(const v8bf*)(p + 8);
  v16bf o;
#pragma unroll
  for (int i = 0; i < 8; ++i) { o[i] = lo[i]; o[8 + i] = hi[i]; }
  return o;
}

// B fragment straight from a row-major fp32 weight matrix W[out][in]
// (row-major [out][in] == exactly the [col][k] layout WMMA wants).
// Used only in the tiny output-MLP kernel.
static __device__ __forceinline__ v16bf frag_b_gf32(const float* base, int ld,
                                                    int col0, int k0, int lane) {
  const int h = (lane >> 4) & 1, c = lane & 15;
  const float* p = base + (size_t)(col0 + c) * ld + k0 + 16 * h;
  v4f x0 = *(const v4f*)p;
  v4f x1 = *(const v4f*)(p + 4);
  v4f x2 = *(const v4f*)(p + 8);
  v4f x3 = *(const v4f*)(p + 12);
  v16bf o;
#pragma unroll
  for (int i = 0; i < 4; ++i) {
    o[i]      = (bf16)x0[i];
    o[4 + i]  = (bf16)x1[i];
    o[8 + i]  = (bf16)x2[i];
    o[12 + i] = (bf16)x3[i];
  }
  return o;
}

// ---------------- workspace zero ----------------
__global__ void k_zero(float* p, int n) {
  int i = blockIdx.x * blockDim.x + threadIdx.x;
  if (i < n) p[i] = 0.0f;
}

// ---------------- fused edge pipeline ----------------
// grid = (2*CHUNKS, B); block = 256 (8 waves).
// blockIdx.x bit0 selects branch (idx_s -> fc1/fc2, idx_a -> fc3/fc4).
// rel_rec tiles are DMA'd global->LDS by the TDM (double-buffered, tracked
// with TENSORcnt) while WMMA works on the current tile; rel_send flows
// through VMEM with global_prefetch_b8 one tile ahead.  All operands are
// converted to bf16 exactly once, so the inner GEMM loops are pure
// ds_load_b128 + v_wmma.  Per 64-edge tile:
//   P  = rel_send_tile @ inputs[b]            (64x64,  K=128)
//   H  = relu(P @ W1^T + b1)                  (64x128, K=64)
//   Mt = (relu(H @ W2^T + b2) * rel_type)^T   (stored [m][e] for B-frags)
//   acc += rel_rec_tile^T @ Mt                (rank-64 update of agg[b])
// acc lives in per-wave v8f[8] across the whole chunk, flushed via f32 atomics.
__global__ __launch_bounds__(256) void k_edge(
    const float* __restrict__ inputs, const float* __restrict__ rel_rec,
    const float* __restrict__ rel_send, const float* __restrict__ rel_type,
    const float* __restrict__ w1s, const float* __restrict__ b1s,
    const float* __restrict__ w2s, const float* __restrict__ b2s,
    const float* __restrict__ w1a, const float* __restrict__ b1a,
    const float* __restrict__ w2a, const float* __restrict__ b2a,
    float* __restrict__ agg) {
  extern __shared__ char smem[];
  bf16* XT  = (bf16*)smem;          // [64][136]   inputs[b]^T  ([col=d][k=n])
  bf16* RS  = XT + 64 * 136;        // [64][136]   rel_send tile (A row-major)
  bf16* P   = RS + 64 * 136;        // [64][72]    pre_msg tile (A row-major)
  bf16* Hs  = P + 64 * 72;          // [64][136]   hidden tile  (A row-major)
  bf16* Mt  = Hs + 64 * 136;        // [128][72]   msg tile transposed [m][e]
  bf16* RRT = Mt + 128 * 72;        // [128][72]   rel_rec^T tile [n][e]
  bf16* W1L = RRT + 128 * 72;       // [2][128][72]  fc1 weights [t][h][d]
  bf16* W2L = W1L + 2 * 128 * 72;   // [2][128][136] fc2 weights [t][m][h]
  float* SRR = (float*)(W2L + 2 * 128 * 136);  // [2][64][128] TDM fp32 staging
  float* rt  = SRR + 2 * ET * NN;              // [64][2] rel_type slice

  const int b = blockIdx.y;
  const int halfIdx = blockIdx.x & 1;
  const int chunk = blockIdx.x >> 1;
  const int tid = threadIdx.x;
  const int wv = tid >> 5, lane = tid & 31;

  const float* w1 = halfIdx ? w1a : w1s;
  const float* b1 = halfIdx ? b1a : b1s;
  const float* w2 = halfIdx ? w2a : w2s;
  const float* b2 = halfIdx ? b2a : b2s;

  const unsigned srrOff =
      (unsigned)__builtin_amdgcn_groupstaticsize() + SRR_BYTE_OFF;

  const int tBeg = chunk * TPC;
  const int tEnd = (tBeg + TPC < NTILE) ? (tBeg + TPC) : NTILE;

  // Kick off the TDM for the first tile's rel_rec block immediately.
  if (wv == 0)
    tdm_load_tile_f32(rel_rec + (size_t)(halfIdx * EHALF + tBeg * ET) * NN,
                      srrOff);

  // ---- one-time staging: inputs[b]^T and this branch's weights, as bf16 ----
  const float* Xb = inputs + (size_t)b * NN * DD;
  for (int idx = tid; idx < NN * DD; idx += 256) {
    int n = idx / DD, d = idx % DD;
    XT[d * 136 + n] = (bf16)Xb[idx];
  }
  for (int idx = tid; idx < TT * HH * DD; idx += 256) {
    int t = idx / (HH * DD), h = (idx / DD) % HH, d = idx % DD;
    W1L[(t * HH + h) * 72 + d] = (bf16)w1[idx];
  }
  for (int idx = tid; idx < TT * MM * HH; idx += 256) {
    int t = idx / (MM * HH), m = (idx / HH) % MM, h = idx % HH;
    W2L[(t * MM + m) * 136 + h] = (bf16)w2[idx];
  }

  v8f acc[8];
#pragma unroll
  for (int i = 0; i < 8; ++i) acc[i] = zero8();
  __syncthreads();

  for (int tile = tBeg; tile < tEnd; ++tile) {
    const int e0 = halfIdx * EHALF + tile * ET;
    const int buf = (tile - tBeg) & 1;

    // Stage rel_send tile (A row-major bf16) via VMEM; coalesced fp32 reads.
    for (int idx = tid; idx < ET * NN; idx += 256) {
      int el = idx / NN, n = idx % NN;
      RS[el * 136 + n] = (bf16)rel_send[(size_t)(e0 + el) * NN + n];
    }
    for (int idx = tid; idx < ET * TT; idx += 256)
      rt[idx] = rel_type[((size_t)b * EE + (e0 + idx / TT)) * TT + (idx % TT)];

    // Pipeline the TDM: issue tile+1's DMA, then wait until only it is
    // outstanding (the oldest -- this tile's buffer -- has landed in LDS).
    // s_wait_tensorcnt takes a SIMM16 immediate, so both paths are literal.
    if (wv == 0) {
      if (tile + 1 < tEnd) {
        tdm_load_tile_f32(rel_rec + (size_t)(e0 + ET) * NN,
                          srrOff + (unsigned)(1 - buf) * SRR_TILE_BYTES);
        __builtin_amdgcn_s_wait_tensorcnt(1);
      } else {
        __builtin_amdgcn_s_wait_tensorcnt(0);
      }
    }
    __syncthreads();

    // LDS->LDS: transpose + convert this tile's rel_rec block to bf16
    // (RRT[n][e_local], the A-matrix of the agg GEMM).
    {
      const float* S = SRR + buf * (ET * NN);
      for (int idx = tid; idx < ET * NN; idx += 256) {
        int el = idx / NN, n = idx % NN;
        RRT[n * 72 + el] = (bf16)S[el * NN + n];
      }
    }

    // Prefetch next tile's rel_send into cache (global_prefetch_b8):
    // 256 threads x 128B covers the full 64x128 fp32 tile.
    if (tile + 1 < tEnd)
      __builtin_prefetch(rel_send + (size_t)(e0 + ET) * NN + (size_t)tid * 32,
                         0, 0);
    __syncthreads();

    // GEMM1: P(64x64) = rel_send_tile @ inputs[b]   (K = N = 128)
#pragma unroll
    for (int i = 0; i < 2; ++i) {
      const int idx = wv * 2 + i, ti = idx >> 2, tj = idx & 3;
      v8f c = zero8();
#pragma unroll
      for (int kk = 0; kk < 4; ++kk) {
        v16bf a = frag_a_lds(RS, 136, ti * 16, kk * 32, lane);
        v16bf bm = frag_b_lds(XT, 136, tj * 16, kk * 32, lane);
        c = wmma_bf16(a, bm, c);
      }
      const int col = tj * 16 + (lane & 15);
      const int rbase = ti * 16 + ((lane >> 4) << 3);
#pragma unroll
      for (int rr = 0; rr < 8; ++rr) P[(rbase + rr) * 72 + col] = (bf16)c[rr];
    }
    __syncthreads();

    for (int t = 0; t < TT; ++t) {
      const bf16* w1t = W1L + t * HH * 72;
      const bf16* w2t = W2L + t * MM * 136;
      const float* b1t = b1 + t * HH;
      const float* b2t = b2 + t * MM;

      // GEMM2: H(64x128) = relu(P @ W1^T + b1)   (K = D = 64)
#pragma unroll
      for (int i = 0; i < 4; ++i) {
        const int idx = wv * 4 + i, ti = idx >> 3, tj = idx & 7;
        v8f c = zero8();
#pragma unroll
        for (int kk = 0; kk < 2; ++kk) {
          v16bf a = frag_a_lds(P, 72, ti * 16, kk * 32, lane);
          v16bf bm = frag_b_lds(w1t, 72, tj * 16, kk * 32, lane);
          c = wmma_bf16(a, bm, c);
        }
        const int col = tj * 16 + (lane & 15);
        const float bias = b1t[col];
        const int rbase = ti * 16 + ((lane >> 4) << 3);
#pragma unroll
        for (int rr = 0; rr < 8; ++rr) {
          float v = c[rr] + bias;
          Hs[(rbase + rr) * 136 + col] = (bf16)(v > 0.f ? v : 0.f);
        }
      }
      __syncthreads();

      // GEMM3: M = relu(H @ W2^T + b2) * rel_type, stored transposed Mt[m][e]
#pragma unroll
      for (int i = 0; i < 4; ++i) {
        const int idx = wv * 4 + i, ti = idx >> 3, tj = idx & 7;
        v8f c = zero8();
#pragma unroll
        for (int kk = 0; kk < 4; ++kk) {
          v16bf a = frag_a_lds(Hs, 136, ti * 16, kk * 32, lane);
          v16bf bm = frag_b_lds(w2t, 136, tj * 16, kk * 32, lane);
          c = wmma_bf16(a, bm, c);
        }
        const int col = tj * 16 + (lane & 15);
        const float bias = b2t[col];
        const int rbase = ti * 16 + ((lane >> 4) << 3);
#pragma unroll
        for (int rr = 0; rr < 8; ++rr) {
          const int e = rbase + rr;
          float v = c[rr] + bias;
          v = (v > 0.f ? v : 0.f) * rt[e * 2 + t];
          Mt[col * 72 + e] = (bf16)v;
        }
      }
      __syncthreads();

      // AGG: acc(16x128 strip of agg[b]) += RRT(row strip wv) @ Mt  (K=64)
      {
        v16bf a0 = frag_a_lds(RRT, 72, wv * 16, 0, lane);
        v16bf a1 = frag_a_lds(RRT, 72, wv * 16, 32, lane);
#pragma unroll
        for (int tj = 0; tj < 8; ++tj) {
          v16bf b0 = frag_b_lds(Mt, 72, tj * 16, 0, lane);
          acc[tj] = wmma_bf16(a0, b0, acc[tj]);
          v16bf b1f = frag_b_lds(Mt, 72, tj * 16, 32, lane);
          acc[tj] = wmma_bf16(a1, b1f, acc[tj]);
        }
      }
      __syncthreads();
    }
  }

  // Flush the chunk's agg contribution: one f32 atomic per element.
  float* aggb = agg + (size_t)b * NN * MM;
  const int col = lane & 15;
  const int rbase = wv * 16 + ((lane >> 4) << 3);
#pragma unroll
  for (int tj = 0; tj < 8; ++tj)
#pragma unroll
    for (int rr = 0; rr < 8; ++rr)
      unsafeAtomicAdd(&aggb[(size_t)(rbase + rr) * MM + tj * 16 + col],
                      acc[tj][rr]);
}

// ---------------- output MLP ----------------
// grid = B; block = 256. agg(128x128) -> relu fc1(256) -> relu fc2(256)
// -> fc3(64) + inputs.  ~1% of total FLOPs; weights consumed as B-fragments
// straight from global (L2-resident).
__global__ __launch_bounds__(256) void k_out(
    const float* __restrict__ agg, const float* __restrict__ inputs,
    const float* __restrict__ w1, const float* __restrict__ b1,
    const float* __restrict__ w2, const float* __restrict__ b2,
    const float* __restrict__ w3, const float* __restrict__ b3,
    float* __restrict__ out) {
  extern __shared__ char smem[];
  bf16* A0 = (bf16*)smem;        // [128][264] (agg, then h2)
  bf16* A1 = A0 + 128 * 264;     // [128][264] (h1)

  const int b = blockIdx.x;
  const int tid = threadIdx.x, wv = tid >> 5, lane = tid & 31;

  const float* aggb = agg + (size_t)b * NN * MM;
  for (int idx = tid; idx < NN * MM; idx += 256) {
    int n = idx / MM, m = idx % MM;
    A0[n * 264 + m] = (bf16)aggb[idx];
  }
  __syncthreads();

  // h1 = relu(agg @ ow1^T + ob1): 128x256, K=128
#pragma unroll
  for (int i = 0; i < 16; ++i) {
    const int idx = wv * 16 + i, ti = idx >> 4, tj = idx & 15;
    v8f c = zero8();
#pragma unroll
    for (int kk = 0; kk < 4; ++kk) {
      v16bf a = frag_a_lds(A0, 264, ti * 16, kk * 32, lane);
      v16bf bm = frag_b_gf32(w1, MM, tj * 16, kk * 32, lane);
      c = wmma_bf16(a, bm, c);
    }
    const int col = tj * 16 + (lane & 15);
    const float bias = b1[col];
    const int rbase = ti * 16 + ((lane >> 4) << 3);
#pragma unroll
    for (int rr = 0; rr < 8; ++rr) {
      float v = c[rr] + bias;
      A1[(rbase + rr) * 264 + col] = (bf16)(v > 0.f ? v : 0.f);
    }
  }
  __syncthreads();

  // h2 = relu(h1 @ ow2^T + ob2): 128x256, K=256
#pragma unroll
  for (int i = 0; i < 16; ++i) {
    const int idx = wv * 16 + i, ti = idx >> 4, tj = idx & 15;
    v8f c = zero8();
#pragma unroll
    for (int kk = 0; kk < 8; ++kk) {
      v16bf a = frag_a_lds(A1, 264, ti * 16, kk * 32, lane);
      v16bf bm = frag_b_gf32(w2, NHH, tj * 16, kk * 32, lane);
      c = wmma_bf16(a, bm, c);
    }
    const int col = tj * 16 + (lane & 15);
    const float bias = b2[col];
    const int rbase = ti * 16 + ((lane >> 4) << 3);
#pragma unroll
    for (int rr = 0; rr < 8; ++rr) {
      float v = c[rr] + bias;
      A0[(rbase + rr) * 264 + col] = (bf16)(v > 0.f ? v : 0.f);
    }
  }
  __syncthreads();

  // pred = h2 @ ow3^T + ob3 ; out = inputs + pred : 128x64, K=256
  const float* Xb = inputs + (size_t)b * NN * DD;
  float* Ob = out + (size_t)b * NN * DD;
#pragma unroll
  for (int i = 0; i < 4; ++i) {
    const int idx = wv * 4 + i, ti = idx >> 2, tj = idx & 3;
    v8f c = zero8();
#pragma unroll
    for (int kk = 0; kk < 8; ++kk) {
      v16bf a = frag_a_lds(A0, 264, ti * 16, kk * 32, lane);
      v16bf bm = frag_b_gf32(w3, NHH, tj * 16, kk * 32, lane);
      c = wmma_bf16(a, bm, c);
    }
    const int col = tj * 16 + (lane & 15);
    const float bias = b3[col];
    const int rbase = ti * 16 + ((lane >> 4) << 3);
#pragma unroll
    for (int rr = 0; rr < 8; ++rr) {
      const int n = rbase + rr;
      Ob[(size_t)n * DD + col] = Xb[(size_t)n * DD + col] + c[rr] + bias;
    }
  }
}

// ---------------- launch ----------------
extern "C" void kernel_launch(void* const* d_in, const int* in_sizes, int n_in,
                              void* d_out, int out_size, void* d_ws,
                              size_t ws_size, hipStream_t stream) {
  const float* inputs   = (const float*)d_in[0];
  const float* rel_rec  = (const float*)d_in[1];
  const float* rel_send = (const float*)d_in[2];
  const float* rel_type = (const float*)d_in[3];
  // d_in[4] = indices: arange(E).reshape(2, E/2) -> contiguous halves, unused.
  const float* w1s = (const float*)d_in[5];
  const float* b1s = (const float*)d_in[6];
  const float* w2s = (const float*)d_in[7];
  const float* b2s = (const float*)d_in[8];
  const float* w1a = (const float*)d_in[9];
  const float* b1a = (const float*)d_in[10];
  const float* w2a = (const float*)d_in[11];
  const float* b2a = (const float*)d_in[12];
  const float* ow1 = (const float*)d_in[13];
  const float* ob1 = (const float*)d_in[14];
  const float* ow2 = (const float*)d_in[15];
  const float* ob2 = (const float*)d_in[16];
  const float* ow3 = (const float*)d_in[17];
  const float* ob3 = (const float*)d_in[18];

  float* agg = (float*)d_ws;  // B*N*M fp32 = 2 MB
  float* out = (float*)d_out;

  const int aggElems = BB * NN * MM;
  k_zero<<<(aggElems + 255) / 256, 256, 0, stream>>>(agg, aggElems);

  // bf16 region (= SRR_BYTE_OFF bytes) + fp32 TDM staging + rel_type slice
  const size_t lds1 = (size_t)SRR_BYTE_OFF +
                      (size_t)2 * SRR_TILE_BYTES +
                      (size_t)ET * TT * sizeof(float);
  k_edge<<<dim3(2 * CHUNKS, BB), 256, lds1, stream>>>(
      inputs, rel_rec, rel_send, rel_type, w1s, b1s, w2s, b2s, w1a, b1a, w2a,
      b2a, agg);

  const size_t lds2 = (size_t)2 * 128 * 264 * sizeof(bf16);
  k_out<<<BB, 256, lds2, stream>>>(agg, inputs, ow1, ob1, ow2, ob2, ow3, ob3,
                                   out);
}
